// TransformerBlockWithLayerSkip_6880537608276
// MI455X (gfx1250) — matrix operand order repaired and checked by reading, not compiled, not run on hardware
//
#include <hip/hip_runtime.h>
#include <cstdint>

typedef _Float16 v16h __attribute__((ext_vector_type(16)));
typedef _Float16 v8h  __attribute__((ext_vector_type(8)));
typedef _Float16 v4h  __attribute__((ext_vector_type(4)));
typedef float    v8f  __attribute__((ext_vector_type(8)));
typedef float    v4f  __attribute__((ext_vector_type(4)));

#define B_  8
#define S_  1024
#define D_  1024
#define H_  16
#define DH_ 64
#define FF_ 4096
#define E_  8
#define T_  (B_*S_)

__device__ __forceinline__ v8f wmma16(v16h a, v16h b, v8f c) {
  // D = A(16x32 f16) x B(32x16 f16) + C(16x16 f32)
  return __builtin_amdgcn_wmma_f32_16x16x32_f16(false, a, false, b, (short)0, c, false, false);
}

__device__ __forceinline__ float gelu_tanh(float x) {
  const float c0 = 0.7978845608028654f; // sqrt(2/pi)
  float x3 = x * x * x;
  return 0.5f * x * (1.0f + tanhf(c0 * (x + 0.044715f * x3)));
}

// ---------- f32 -> f16 convert (layout preserved, vectorized x4) -------
__global__ __launch_bounds__(256) void cvt_kernel(
    const float* __restrict__ s, _Float16* __restrict__ d)
{
  int i = (blockIdx.x * 256 + threadIdx.x) * 4;
  v4f v = *(const v4f*)(s + i);
  v4h o;
  #pragma unroll
  for (int j = 0; j < 4; ++j) o[j] = (_Float16)v[j];
  *(v4h*)(d + i) = o;
}

// ---------- f32 [K,N] -> f16 [N,K] transpose-convert (32x32 tiles) -----
__global__ __launch_bounds__(256) void tcvt_kernel(
    const float* __restrict__ src, _Float16* __restrict__ dst, int K, int N)
{
  __shared__ _Float16 tile[32][33];
  int k0 = blockIdx.y * 32;
  int n0 = blockIdx.x * 32;
  int tx = threadIdx.x & 31;
  int ty = threadIdx.x >> 5;
  #pragma unroll
  for (int r = ty; r < 32; r += 8)
    tile[r][tx] = (_Float16)src[(size_t)(k0 + r) * N + n0 + tx];
  __syncthreads();
  #pragma unroll
  for (int r = ty; r < 32; r += 8)
    dst[(size_t)(n0 + r) * K + k0 + tx] = tile[tx][r];
}

// ---------------- LayerNorm (block per token) -> f16 -------------------
__global__ __launch_bounds__(256) void ln_kernel(
    const float* __restrict__ x, const float* __restrict__ g,
    const float* __restrict__ be, _Float16* __restrict__ o)
{
  __shared__ float red[256];
  int t = blockIdx.x, tid = threadIdx.x;
  const float* row = x + (size_t)t * D_;
  float vals[4]; float s = 0.f;
  #pragma unroll
  for (int i = 0; i < 4; ++i) { vals[i] = row[tid + i * 256]; s += vals[i]; }
  red[tid] = s; __syncthreads();
  for (int off = 128; off > 0; off >>= 1) { if (tid < off) red[tid] += red[tid + off]; __syncthreads(); }
  float mean = red[0] * (1.0f / D_);
  __syncthreads();
  float vv = 0.f;
  #pragma unroll
  for (int i = 0; i < 4; ++i) { float d = vals[i] - mean; vv += d * d; }
  red[tid] = vv; __syncthreads();
  for (int off = 128; off > 0; off >>= 1) { if (tid < off) red[tid] += red[tid + off]; __syncthreads(); }
  float rstd = rsqrtf(red[0] * (1.0f / D_) + 1e-5f);
  #pragma unroll
  for (int i = 0; i < 4; ++i) {
    int d = tid + i * 256;
    o[(size_t)t * D_ + d] = (_Float16)((vals[i] - mean) * rstd * g[d] + be[d]);
  }
}

// ------------- Fused LN2 + router softmax/top-2 combine weights --------
__global__ __launch_bounds__(256) void ln2_router_kernel(
    const float* __restrict__ x, const float* __restrict__ g,
    const float* __restrict__ be, const float* __restrict__ rw,
    _Float16* __restrict__ ln2, float* __restrict__ cw)
{
  __shared__ float red[256];
  __shared__ float wsum[8][E_];
  int t = blockIdx.x, tid = threadIdx.x;
  const float* row = x + (size_t)t * D_;
  float vals[4]; float s = 0.f;
  #pragma unroll
  for (int i = 0; i < 4; ++i) { vals[i] = row[tid + i * 256]; s += vals[i]; }
  red[tid] = s; __syncthreads();
  for (int off = 128; off > 0; off >>= 1) { if (tid < off) red[tid] += red[tid + off]; __syncthreads(); }
  float mean = red[0] * (1.0f / D_);
  __syncthreads();
  float vv = 0.f;
  #pragma unroll
  for (int i = 0; i < 4; ++i) { float d = vals[i] - mean; vv += d * d; }
  red[tid] = vv; __syncthreads();
  for (int off = 128; off > 0; off >>= 1) { if (tid < off) red[tid] += red[tid + off]; __syncthreads(); }
  float rstd = rsqrtf(red[0] * (1.0f / D_) + 1e-5f);

  float acc[E_];
  #pragma unroll
  for (int e = 0; e < E_; ++e) acc[e] = 0.f;
  #pragma unroll
  for (int i = 0; i < 4; ++i) {
    int d = tid + i * 256;
    float nv = (vals[i] - mean) * rstd * g[d] + be[d];
    ln2[(size_t)t * D_ + d] = (_Float16)nv;
    #pragma unroll
    for (int e = 0; e < E_; ++e) acc[e] += nv * rw[(size_t)d * E_ + e];
  }
  #pragma unroll
  for (int e = 0; e < E_; ++e)
    for (int m = 16; m > 0; m >>= 1) acc[e] += __shfl_xor(acc[e], m);
  int wave = tid >> 5, lane = tid & 31;
  if (lane == 0) {
    #pragma unroll
    for (int e = 0; e < E_; ++e) wsum[wave][e] = acc[e];
  }
  __syncthreads();
  if (tid == 0) {
    float lg[E_];
    for (int e = 0; e < E_; ++e) { float q = 0.f; for (int w = 0; w < 8; ++w) q += wsum[w][e]; lg[e] = q; }
    float mx = lg[0];
    for (int e = 1; e < E_; ++e) mx = fmaxf(mx, lg[e]);
    float pe[E_], ps = 0.f;
    for (int e = 0; e < E_; ++e) { pe[e] = __expf(lg[e] - mx); ps += pe[e]; }
    for (int e = 0; e < E_; ++e) pe[e] /= ps;
    int i1 = 0;
    for (int e = 1; e < E_; ++e) if (pe[e] > pe[i1]) i1 = e;
    int i2 = (i1 == 0) ? 1 : 0;
    for (int e = 0; e < E_; ++e) if (e != i1 && pe[e] > pe[i2]) i2 = e;
    float den = pe[i1] + pe[i2];
    float o[E_];
    for (int e = 0; e < E_; ++e) o[e] = 0.f;
    o[i1] = pe[i1] / den; o[i2] = pe[i2] / den;
    for (int e = 0; e < E_; ++e) cw[(size_t)t * E_ + e] = o[e];
  }
}

// ------- WMMA GEMM: out = A(f16,[M,K]) x B(f16,[N,K] K-major) ----------
// EPI 0: +bias, store f16 (+ optional V-transpose scatter for QKV)
// EPI 1: +bias+residual, store f32
// EPI 2: +bias, GELU, store f16 (top-2 tile skip)
// EPI 3: out += cw*(acc+bias), f32 (top-2 tile skip)
template<int EPI>
__global__ __launch_bounds__(256) void gemm_kernel(
    const _Float16* __restrict__ A, int lda,
    const _Float16* __restrict__ Bh, int ldb,
    const float* __restrict__ bias,
    _Float16* __restrict__ outH, float* __restrict__ outF,
    const float* __restrict__ residual,
    _Float16* __restrict__ vt,
    const float* __restrict__ cw, int expert,
    int N, int Kd)
{
  if constexpr (EPI >= 2) {
    __shared__ int flag;
    if (threadIdx.x == 0) flag = 0;
    __syncthreads();
    if (threadIdx.x < 32) {
      int t = blockIdx.y * 32 + threadIdx.x;
      if (cw[(size_t)t * E_ + expert] > 0.0f) flag = 1;
    }
    __syncthreads();
    if (!flag) return;  // block-uniform: EXEC stays all-1s for WMMA
  }
  int lane = threadIdx.x & 31;
  int wave = threadIdx.x >> 5;
  int wm = wave >> 2, wn = wave & 3;
  int lh = lane >> 4, ll = lane & 15;
  int m0 = blockIdx.y * 32 + wm * 16;
  int n0 = blockIdx.x * 256 + wn * 64;

  v8f c[4] = {};
  for (int k0 = 0; k0 < Kd; k0 += 32) {
    // A tile 16x32: lane half lh holds K = {8lh..8lh+7, 16+8lh..16+8lh+7}
    const _Float16* ap = A + (size_t)(m0 + ll) * lda + k0 + lh * 8;
    v8h a_lo = *(const v8h*)ap;
    v8h a_hi = *(const v8h*)(ap + 16);
    v16h a;
    #pragma unroll
    for (int i = 0; i < 8; ++i) { a[i] = a_lo[i]; a[8 + i] = a_hi[i]; }
    int kb = k0 + lh * 8;       // B: lane half holds K = {8lh..8lh+7, 16+8lh..16+8lh+7}
    #pragma unroll
    for (int nt = 0; nt < 4; ++nt) {
      int n = n0 + nt * 16 + ll;
      const _Float16* bp = Bh + (size_t)n * ldb + kb;
      v8h b_lo = *(const v8h*)bp;
      v8h b_hi = *(const v8h*)(bp + 16);
      v16h b;
      #pragma unroll
      for (int i = 0; i < 8; ++i) { b[i] = b_lo[i]; b[8 + i] = b_hi[i]; }
      c[nt] = wmma16(a, b, c[nt]);
    }
  }
  #pragma unroll
  for (int nt = 0; nt < 4; ++nt) {
    int n = n0 + nt * 16 + ll;
    float bv = bias[n];
    #pragma unroll
    for (int r = 0; r < 8; ++r) {
      int m = m0 + lh * 8 + r;   // C/D layout: M = r + 8*(lane/16)
      size_t idx = (size_t)m * N + n;
      float v = c[nt][r] + bv;
      if constexpr (EPI == 0) {
        outH[idx] = (_Float16)v;
        if (n >= 2 * D_) {       // scatter V into [B,H,DH,S] for P.V loads
          int dh = (n - 2 * D_) & (DH_ - 1);
          int hh = (n - 2 * D_) >> 6;
          int bb = m >> 10, ss = m & (S_ - 1);
          vt[((size_t)(bb * H_ + hh) * DH_ + dh) * S_ + ss] = (_Float16)v;
        }
      }
      else if constexpr (EPI == 1) outF[idx] = v + residual[idx];
      else if constexpr (EPI == 2) outH[idx] = (_Float16)gelu_tanh(v);
      else {
        float w = cw[(size_t)m * E_ + expert];
        if (w > 0.f) outF[idx] += w * v;
      }
    }
  }
}

// ---------------- Attention: 1 block (4 waves) per 16-query tile -------
__global__ __launch_bounds__(128) void attn_kernel(
    const _Float16* __restrict__ qkv, const _Float16* __restrict__ vt,
    _Float16* __restrict__ attn)
{
  __shared__ float sc[16][S_];   // 64 KB: full score rows for 16 queries
  int lane = threadIdx.x & 31;
  int wave = threadIdx.x >> 5;   // 0..3
  int lh = lane >> 4, ll = lane & 15;
  int qt = blockIdx.x & (S_ / 16 - 1);
  int bh = blockIdx.x >> 6;
  int h = bh & (H_ - 1);
  int b = bh >> 4;

  // Q tile as A matrices for K-steps 0..31 and 32..63 of DH
  v16h a0, a1;
  {
    size_t tq = (size_t)b * S_ + qt * 16 + ll;
    const _Float16* p = qkv + tq * (3 * D_) + h * DH_ + lh * 8;
    v8h x0 = *(const v8h*)p,        x1 = *(const v8h*)(p + 16);
    v8h x2 = *(const v8h*)(p + 32), x3 = *(const v8h*)(p + 48);
    #pragma unroll
    for (int i = 0; i < 8; ++i) { a0[i] = x0[i]; a0[8 + i] = x1[i]; a1[i] = x2[i]; a1[8 + i] = x3[i]; }
  }
  // scores = Q K^T / 8 ; waves split the 64 kv tiles
  for (int j = wave; j < S_ / 16; j += 4) {
    size_t tk = (size_t)b * S_ + j * 16 + ll;
    const _Float16* p = qkv + tk * (3 * D_) + D_ + h * DH_ + lh * 8;
    v8h x0 = *(const v8h*)p,        x1 = *(const v8h*)(p + 16);
    v8h x2 = *(const v8h*)(p + 32), x3 = *(const v8h*)(p + 48);
    v16h b0, b1;
    #pragma unroll
    for (int i = 0; i < 8; ++i) { b0[i] = x0[i]; b0[8 + i] = x1[i]; b1[i] = x2[i]; b1[8 + i] = x3[i]; }
    v8f c = {};
    c = wmma16(a0, b0, c);
    c = wmma16(a1, b1, c);
    #pragma unroll
    for (int r = 0; r < 8; ++r) sc[lh * 8 + r][j * 16 + ll] = c[r] * 0.125f;
  }
  __syncthreads();
  // softmax: 8 threads per row, float4-vectorized, wave32 shuffle reduce
  {
    int row = threadIdx.x >> 3;
    v4f* rp = (v4f*)&sc[row][(threadIdx.x & 7) * 128];
    float mx = -3.0e38f;
    #pragma unroll 8
    for (int i = 0; i < 32; ++i) {
      v4f q = rp[i];
      mx = fmaxf(mx, fmaxf(fmaxf(q[0], q[1]), fmaxf(q[2], q[3])));
    }
    #pragma unroll
    for (int m = 1; m < 8; m <<= 1) mx = fmaxf(mx, __shfl_xor(mx, m));
    float sum = 0.f;
    #pragma unroll 8
    for (int i = 0; i < 32; ++i) {
      v4f q = rp[i];
      #pragma unroll
      for (int j = 0; j < 4; ++j) { q[j] = __expf(q[j] - mx); sum += q[j]; }
      rp[i] = q;
    }
    #pragma unroll
    for (int m = 1; m < 8; m <<= 1) sum += __shfl_xor(sum, m);
    float inv = 1.0f / sum;
    #pragma unroll 8
    for (int i = 0; i < 32; ++i) {
      v4f q = rp[i];
      #pragma unroll
      for (int j = 0; j < 4; ++j) q[j] *= inv;
      rp[i] = q;
    }
  }
  __syncthreads();
  // O = P V ; wave computes dh tile nt = wave (16 cols), K = 1024 kv
  int nt = wave;
  v8f o = {};
  for (int kk = 0; kk < S_ / 32; ++kk) {
    v16h a;
    {
      int kb = kk * 32 + lh * 8;
      v4f p0 = *(const v4f*)&sc[ll][kb];
      v4f p1 = *(const v4f*)&sc[ll][kb + 4];
      v4f p2 = *(const v4f*)&sc[ll][kb + 16];
      v4f p3 = *(const v4f*)&sc[ll][kb + 20];
      #pragma unroll
      for (int i = 0; i < 4; ++i) {
        a[i] = (_Float16)p0[i];      a[4 + i]  = (_Float16)p1[i];
        a[8 + i] = (_Float16)p2[i];  a[12 + i] = (_Float16)p3[i];
      }
    }
    // V from [B,H,DH,S]: contiguous over kv
    const _Float16* vp = vt + ((size_t)bh * DH_ + nt * 16 + ll) * S_ + kk * 32 + lh * 8;
    v8h y0 = *(const v8h*)vp;
    v8h y1 = *(const v8h*)(vp + 16);
    v16h bv;
    #pragma unroll
    for (int i = 0; i < 8; ++i) { bv[i] = y0[i]; bv[8 + i] = y1[i]; }
    o = wmma16(a, bv, o);
  }
  #pragma unroll
  for (int r = 0; r < 8; ++r) {
    size_t to = (size_t)b * S_ + qt * 16 + lh * 8 + r;
    attn[to * D_ + h * DH_ + nt * 16 + ll] = (_Float16)o[r];
  }
}

__global__ __launch_bounds__(256) void zero_kernel(float* __restrict__ p, int n) {
  int i = blockIdx.x * 256 + threadIdx.x;
  if (i < n) p[i] = 0.f;
}

extern "C" void kernel_launch(void* const* d_in, const int* in_sizes, int n_in,
                              void* d_out, int out_size, void* d_ws, size_t ws_size,
                              hipStream_t stream)
{
  (void)in_sizes; (void)n_in; (void)out_size; (void)ws_size;
  const float* hs   = (const float*)d_in[0];
  const float* inw  = (const float*)d_in[1];
  const float* inb  = (const float*)d_in[2];
  const float* outw = (const float*)d_in[3];
  const float* outb = (const float*)d_in[4];
  const float* ln1g = (const float*)d_in[5];
  const float* ln1b = (const float*)d_in[6];
  const float* ln2g = (const float*)d_in[7];
  const float* ln2b = (const float*)d_in[8];
  const float* rw   = (const float*)d_in[9];
  const float* w1   = (const float*)d_in[10];
  const float* b1   = (const float*)d_in[11];
  const float* w2   = (const float*)d_in[12];
  const float* b2   = (const float*)d_in[13];
  float* out = (float*)d_out;

  char* ws = (char*)d_ws;
  _Float16* ln1f   = (_Float16*)(ws);                  // 16 MiB (reused as attn)
  _Float16* qkvf   = (_Float16*)(ws + (16u  << 20));   // 48 MiB
  _Float16* attnf  = ln1f;
  _Float16* hexp   = (_Float16*)(ws);                  // 64 MiB (reuses ln1+qkv)
  float*    h2     = (float*)(ws + (64u  << 20));      // 32 MiB
  _Float16* ln2f   = (_Float16*)(ws + (96u  << 20));   // 16 MiB
  float*    cw     = (float*)(ws + (112u << 20));      // 256 KiB (pad to 1 MiB)
  _Float16* inw16  = (_Float16*)(ws + (113u << 20));   // 6 MiB
  _Float16* outw16 = (_Float16*)(ws + (119u << 20));   // 2 MiB
  _Float16* vt     = (_Float16*)(ws + (121u << 20));   // 16 MiB [B,H,DH,S]
  _Float16* w1t    = (_Float16*)(ws + (137u << 20));   // 8 MiB  (per-expert [FF,D])
  _Float16* w2t    = (_Float16*)(ws + (145u << 20));   // 8 MiB  (per-expert [D,FF])

  // 0) one-time f32->f16 weight mirrors (already K-major: plain convert)
  cvt_kernel<<<(3 * D_ * D_) / 1024, 256, 0, stream>>>(inw,  inw16);
  cvt_kernel<<<(D_ * D_) / 1024,     256, 0, stream>>>(outw, outw16);

  // 1) LN1 -> f16
  ln_kernel<<<T_, 256, 0, stream>>>(hs, ln1g, ln1b, ln1f);
  // 2) QKV = ln1 @ Wi^T + bi ; also scatter V into [B,H,DH,S]
  gemm_kernel<0><<<dim3(3 * D_ / 256, T_ / 32), 256, 0, stream>>>(
      ln1f, D_, inw16, D_, inb, qkvf, nullptr, nullptr, vt, nullptr, 0, 3 * D_, D_);
  // 3) attention
  attn_kernel<<<B_ * H_ * (S_ / 16), 128, 0, stream>>>(qkvf, vt, attnf);
  // 4) h2 = attn @ Wo^T + bo + residual
  gemm_kernel<1><<<dim3(D_ / 256, T_ / 32), 256, 0, stream>>>(
      attnf, D_, outw16, D_, outb, nullptr, h2, hs, nullptr, nullptr, 0, D_, D_);
  // 5) LN2 + router top-2 combine weights
  ln2_router_kernel<<<T_, 256, 0, stream>>>(h2, ln2g, ln2b, rw, ln2f, cw);
  // 6) MoE, top-2 sparse at 32-token tile granularity; per-expert
  //    transpose-convert weights to f16 K-major so GEMM B loads are b128
  zero_kernel<<<(T_ * D_) / 256, 256, 0, stream>>>(out, T_ * D_);
  for (int e = 0; e < E_; ++e) {
    tcvt_kernel<<<dim3(FF_ / 32, D_ / 32), 256, 0, stream>>>(
        w1 + (size_t)e * D_ * FF_, w1t, D_, FF_);          // [D,FF] -> [FF,D]
    gemm_kernel<2><<<dim3(FF_ / 256, T_ / 32), 256, 0, stream>>>(
        ln2f, D_, w1t, D_, b1 + (size_t)e * FF_,
        hexp, nullptr, nullptr, nullptr, cw, e, FF_, D_);
    tcvt_kernel<<<dim3(D_ / 32, FF_ / 32), 256, 0, stream>>>(
        w2 + (size_t)e * FF_ * D_, w2t, FF_, D_);          // [FF,D] -> [D,FF]
    gemm_kernel<3><<<dim3(D_ / 256, T_ / 32), 256, 0, stream>>>(
        hexp, FF_, w2t, FF_, b2 + (size_t)e * D_,
        nullptr, out, nullptr, nullptr, cw, e, D_, FF_);
  }
}
